// Attention_4148938408184
// MI455X (gfx1250) — matrix operand order repaired and checked by reading, not compiled, not run on hardware
//
#include <hip/hip_runtime.h>
#include <hip/hip_bf16.h>

typedef _Float16 f16_t;
typedef __attribute__((ext_vector_type(16))) _Float16 v16h;
typedef __attribute__((ext_vector_type(8)))  _Float16 v8h;
typedef __attribute__((ext_vector_type(8)))  float    v8f;

#define C_ 256
#define NSP 4096
#define BATCH 4
#define GN_G 32
#define WELEM (C_ * C_)  // 65536 elements per weight matrix

#define WMMA_F32_F16(A, B, Cc) \
  __builtin_amdgcn_wmma_f32_16x16x32_f16(false, (A), false, (B), (short)0, (Cc), false, false)

// A-matrix 16x32 f16 fragment from row-major [rows][ld] storage.
// Layout used here: lane m=lane&15; halves i -> K=kb+i (contiguous, matches B frags)
// with kb = 0 (lanes 0-15) or 16 (lanes 16-31).
__device__ inline v16h load_a_frag(const f16_t* M, int ld, int row0, int col0, int lane) {
  int m  = lane & 15;
  int kb = (lane < 16) ? 0 : 16;
  const f16_t* p = M + (size_t)(row0 + m) * ld + col0 + kb;
  v8h lo = *(const v8h*)p;
  v8h hi = *(const v8h*)(p + 8);
  v16h r;
#pragma unroll
  for (int i = 0; i < 8; ++i) { r[i] = lo[i]; r[8 + i] = hi[i]; }
  return r;
}

// B-matrix 32x16 f16 fragment where storage is [n][K] row-major (K contiguous).
// Layout used here: lane col n=lane&15; halves i -> K = k0 + (lane<16?0:16) + i  -> one v16h load.
__device__ inline v16h load_b_frag(const f16_t* M, int ld, int n0, int k0, int lane) {
  int n = n0 + (lane & 15);
  int k = k0 + ((lane < 16) ? 0 : 16);
  return *(const v16h*)(M + (size_t)n * ld + k);
}

// One 16x16 output tile of  D = A(16x256) * W16^T  accumulated over K=256 (8 WMMAs).
__device__ inline v8f gemm_tile(const v16h ax[8], const f16_t* __restrict__ W16,
                                int oc0, int lane) {
  v8f d = {};
#pragma unroll
  for (int ck = 0; ck < 8; ++ck) {
    v16h bw = load_b_frag(W16, C_, oc0, ck * 32, lane);
    d = WMMA_F32_F16(ax[ck], bw, d);
  }
  return d;
}

// ---------------- Kernel 0: one-time f32 -> f16 weight conversion ----------------
__global__ __launch_bounds__(256) void cvt_w_kernel(
    const float* __restrict__ wq, const float* __restrict__ wk,
    const float* __restrict__ wv, const float* __restrict__ wo,
    f16_t* __restrict__ w16) {
  int i = blockIdx.x * 256 + threadIdx.x;  // 0..65535
  w16[i]             = (f16_t)wq[i];
  w16[WELEM + i]     = (f16_t)wk[i];
  w16[2 * WELEM + i] = (f16_t)wv[i];
  w16[3 * WELEM + i] = (f16_t)wo[i];
}

// ---------------- Kernel 1: fused QKV projection ----------------
// outputs: qT[b][n][c] f16, kT[b][n][c] f16, v[b][c][n] f16
__global__ __launch_bounds__(128) void qkv_proj_kernel(
    const float* __restrict__ x, const f16_t* __restrict__ w16,
    const float* __restrict__ bq, const float* __restrict__ bk,
    const float* __restrict__ bvp,
    f16_t* __restrict__ qT, f16_t* __restrict__ kT, f16_t* __restrict__ vmat) {
  __shared__ f16_t xT[16 * 264];  // padded stride
  const int blk  = blockIdx.x;    // b*256 + ntile
  const int b    = blk >> 8;
  const int n0   = (blk & 255) << 4;
  const int tid  = threadIdx.x;
  const int lane = tid & 31;
  const int wave = tid >> 5;

  const float* xb = x + (size_t)b * C_ * NSP;
#pragma unroll 4
  for (int it = 0; it < 32; ++it) {  // stage x^T tile as f16
    int c = (tid >> 4) + it * 8;
    int j = tid & 15;
    xT[j * 264 + c] = (f16_t)xb[(size_t)c * NSP + n0 + j];
  }
  __syncthreads();

  v16h ax[8];
#pragma unroll
  for (int ck = 0; ck < 8; ++ck) ax[ck] = load_a_frag(xT, 264, 0, ck * 32, lane);

  const int row = (lane < 16) ? 0 : 8;
  const int nl  = lane & 15;

  // Q
#pragma unroll
  for (int t = 0; t < 4; ++t) {
    int oc0 = (wave * 4 + t) * 16;
    v8f d = gemm_tile(ax, w16, oc0, lane);
    int oc = oc0 + nl;
    float bb = bq[oc];
    f16_t* dst = qT + ((size_t)b * NSP + n0) * C_ + oc;
#pragma unroll
    for (int r = 0; r < 8; ++r) dst[(size_t)(row + r) * C_] = (f16_t)(d[r] + bb);
  }
  // K
#pragma unroll
  for (int t = 0; t < 4; ++t) {
    int oc0 = (wave * 4 + t) * 16;
    v8f d = gemm_tile(ax, w16 + WELEM, oc0, lane);
    int oc = oc0 + nl;
    float bb = bk[oc];
    f16_t* dst = kT + ((size_t)b * NSP + n0) * C_ + oc;
#pragma unroll
    for (int r = 0; r < 8; ++r) dst[(size_t)(row + r) * C_] = (f16_t)(d[r] + bb);
  }
  // V (stored [c][n] so flash PV B-frags are contiguous)
#pragma unroll
  for (int t = 0; t < 4; ++t) {
    int oc0 = (wave * 4 + t) * 16;
    v8f d = gemm_tile(ax, w16 + 2 * WELEM, oc0, lane);
    int oc = oc0 + nl;
    float bb = bvp[oc];
    f16_t* dst = vmat + ((size_t)b * C_ + oc) * NSP + n0;
#pragma unroll
    for (int r = 0; r < 8; ++r) dst[row + r] = (f16_t)(d[r] + bb);
  }
}

// ---------------- Kernel 2: flash attention (online softmax) ----------------
__global__ __launch_bounds__(128) void flash_attn_kernel(
    const f16_t* __restrict__ qT, const f16_t* __restrict__ kT,
    const f16_t* __restrict__ vmat, f16_t* __restrict__ hT) {
  __shared__ f16_t plds_all[4][16 * 32];
  const int tid  = threadIdx.x;
  const int lane = tid & 31;
  const int wave = tid >> 5;
  f16_t* plds = plds_all[wave];

  const int gw = blockIdx.x * 4 + wave;  // b*256 + qtile
  const int b  = gw >> 8;
  const int n0 = (gw & 255) << 4;

  const f16_t* qTb = qT + (size_t)b * NSP * C_;
  const f16_t* kTb = kT + (size_t)b * NSP * C_;
  const f16_t* vb  = vmat + (size_t)b * C_ * NSP;

  v16h aq[8];
#pragma unroll
  for (int ck = 0; ck < 8; ++ck) aq[ck] = load_a_frag(qTb, C_, n0, ck * 32, lane);

  v8f zero = {};
  v8f acc[16];
#pragma unroll
  for (int t = 0; t < 16; ++t) acc[t] = zero;
  float mx[8], sm[8];
#pragma unroll
  for (int r = 0; r < 8; ++r) { mx[r] = -1e30f; sm[r] = 0.f; }

  const int row = (lane < 16) ? 0 : 8;
  const int nl  = lane & 15;

  for (int kb = 0; kb < NSP; kb += 32) {
    if (kb + 32 < NSP)  // speculative prefetch of next key-block (gfx1250 global_prefetch)
      __builtin_prefetch(kTb + (size_t)(kb + 32 + nl) * C_, 0, 0);

    v8f s0 = zero, s1 = zero;
#pragma unroll
    for (int ck = 0; ck < 8; ++ck) {
      v16h b0 = load_b_frag(kTb, C_, kb, ck * 32, lane);
      v16h b1 = load_b_frag(kTb, C_, kb + 16, ck * 32, lane);
      s0 = WMMA_F32_F16(aq[ck], b0, s0);
      s1 = WMMA_F32_F16(aq[ck], b1, s1);
    }
    float scale[8];
#pragma unroll
    for (int r = 0; r < 8; ++r) {
      float t = fmaxf(s0[r], s1[r]);
#pragma unroll
      for (int d = 1; d < 16; d <<= 1) t = fmaxf(t, __shfl_xor(t, d, 32));
      float nm = fmaxf(mx[r], t);
      scale[r] = __expf(mx[r] - nm);
      mx[r]    = nm;
      float p0 = __expf(s0[r] - nm);
      float p1 = __expf(s1[r] - nm);
      s0[r] = p0; s1[r] = p1;
      float ps = p0 + p1;
#pragma unroll
      for (int d = 1; d < 16; d <<= 1) ps += __shfl_xor(ps, d, 32);
      sm[r] = sm[r] * scale[r] + ps;
    }
#pragma unroll
    for (int t = 0; t < 16; ++t)
#pragma unroll
      for (int r = 0; r < 8; ++r) acc[t][r] *= scale[r];

#pragma unroll
    for (int r = 0; r < 8; ++r) {
      plds[(row + r) * 32 + nl]      = (f16_t)s0[r];
      plds[(row + r) * 32 + 16 + nl] = (f16_t)s1[r];
    }
    v16h ap = load_a_frag(plds, 32, 0, 0, lane);

#pragma unroll
    for (int t = 0; t < 16; ++t) {
      v16h bvf = load_b_frag(vb, NSP, t * 16, kb, lane);
      acc[t] = WMMA_F32_F16(ap, bvf, acc[t]);
    }
  }

  float rinv[8];
#pragma unroll
  for (int r = 0; r < 8; ++r) rinv[r] = 1.0f / sm[r];
  f16_t* dst = hT + ((size_t)b * NSP + n0) * C_;
#pragma unroll
  for (int t = 0; t < 16; ++t) {
    int c = t * 16 + nl;
#pragma unroll
    for (int r = 0; r < 8; ++r)
      dst[(size_t)(row + r) * C_ + c] = (f16_t)(acc[t][r] * rinv[r]);
  }
}

// ---------------- Kernel 3: output projection + bias + residual ----------------
__global__ __launch_bounds__(128) void out_proj_kernel(
    const f16_t* __restrict__ hT, const f16_t* __restrict__ wo16,
    const float* __restrict__ bo, const float* __restrict__ x,
    float* __restrict__ y) {
  const int blk  = blockIdx.x;
  const int b    = blk >> 8;
  const int n0   = (blk & 255) << 4;
  const int tid  = threadIdx.x;
  const int lane = tid & 31;
  const int wave = tid >> 5;

  const f16_t* hb = hT + (size_t)b * NSP * C_;
  v16h ah[8];
#pragma unroll
  for (int ck = 0; ck < 8; ++ck) ah[ck] = load_a_frag(hb, C_, n0, ck * 32, lane);

  const int row = (lane < 16) ? 0 : 8;
  const int nl  = lane & 15;

#pragma unroll
  for (int t = 0; t < 4; ++t) {
    int oc0 = (wave * 4 + t) * 16;
    v8f d = gemm_tile(ah, wo16, oc0, lane);
    int oc = oc0 + nl;
    float bb = bo[oc];
    const float* xr = x + ((size_t)b * C_ + oc) * NSP + n0;
    float*       yr = y + ((size_t)b * C_ + oc) * NSP + n0;
#pragma unroll
    for (int r = 0; r < 8; ++r) yr[row + r] = d[r] + bb + xr[row + r];
  }
}

// ---------------- Kernel 4: GroupNorm + swish ----------------
__global__ __launch_bounds__(256) void gn_swish_kernel(
    const float* __restrict__ y, const float* __restrict__ gamma,
    const float* __restrict__ beta, float* __restrict__ out) {
  const int b = blockIdx.x >> 5;  // 32 groups
  const int g = blockIdx.x & 31;
  const int CPG = C_ / GN_G;      // 8
  const int cnt = CPG * NSP;      // 32768
  const float* base = y + ((size_t)b * C_ + g * CPG) * NSP;
  float s = 0.f, ss = 0.f;
  for (int i = threadIdx.x; i < cnt; i += 256) {
    float v = base[i];
    s += v; ss += v * v;
  }
  __shared__ float rs[256], rss[256];
  rs[threadIdx.x] = s; rss[threadIdx.x] = ss;
  __syncthreads();
  for (int off = 128; off > 0; off >>= 1) {
    if (threadIdx.x < off) {
      rs[threadIdx.x]  += rs[threadIdx.x + off];
      rss[threadIdx.x] += rss[threadIdx.x + off];
    }
    __syncthreads();
  }
  float mu   = rs[0] / (float)cnt;
  float var  = rss[0] / (float)cnt - mu * mu;
  float rsig = rsqrtf(var + 1e-5f);
  float* ob = out + ((size_t)b * C_ + g * CPG) * NSP;
  for (int i = threadIdx.x; i < cnt; i += 256) {
    int c = g * CPG + i / NSP;
    float yn = (base[i] - mu) * rsig * gamma[c] + beta[c];
    ob[i] = yn / (1.f + __expf(-yn));  // swish
  }
}

extern "C" void kernel_launch(void* const* d_in, const int* in_sizes, int n_in,
                              void* d_out, int out_size, void* d_ws, size_t ws_size,
                              hipStream_t stream) {
  const float* x     = (const float*)d_in[0];
  const float* wq    = (const float*)d_in[1];
  const float* bq    = (const float*)d_in[2];
  const float* wk    = (const float*)d_in[3];
  const float* bk    = (const float*)d_in[4];
  const float* wv    = (const float*)d_in[5];
  const float* bv    = (const float*)d_in[6];
  const float* wo    = (const float*)d_in[7];
  const float* bo    = (const float*)d_in[8];
  const float* gamma = (const float*)d_in[9];
  const float* beta  = (const float*)d_in[10];

  const size_t tsz = (size_t)BATCH * NSP * C_;  // elements per f16 tensor
  f16_t* qT  = (f16_t*)d_ws;
  f16_t* kT  = qT + tsz;
  f16_t* vm  = kT + tsz;
  f16_t* hT  = vm + tsz;
  f16_t* w16 = hT + tsz;     // [wq|wk|wv|wo] f16, 512 KB, L2-resident
  float* y   = (float*)d_ws; // reuses qT+kT (16MB) after attention; hT/v/w16 disjoint

  cvt_w_kernel<<<WELEM / 256, 256, 0, stream>>>(wq, wk, wv, wo, w16);
  qkv_proj_kernel<<<BATCH * (NSP / 16), 128, 0, stream>>>(x, w16, bq, bk, bv, qT, kT, vm);
  flash_attn_kernel<<<BATCH * (NSP / 16) / 4, 128, 0, stream>>>(qT, kT, vm, hT);
  out_proj_kernel<<<BATCH * (NSP / 16), 128, 0, stream>>>(hT, w16 + 3 * WELEM, bo, x, y);
  gn_swish_kernel<<<BATCH * GN_G, 256, 0, stream>>>(y, gamma, beta, (float*)d_out);
}
